// Injector_43344809951435
// MI455X (gfx1250) — compile-verified, matrix operand order checked by reading, not run on hardware
//
#include <hip/hip_runtime.h>
#include <hip/hip_bf16.h>

// ---------------- problem constants ----------------
#define DIMC   384
#define HEADS  6
#define DH     64
#define LEVELS 3
#define POINTS 4
#define FFN    1024
#define NQC    4096
#define BC     4
#define NVTOT  21504            // 128*128 + 64*64 + 32*32
#define MQ     (BC * NQC)       // 16384
#define MV     (BC * NVTOT)     // 86016
#define NOA    224              // 144 (off) + 72 (aw) padded to 224 = 7*32

typedef __attribute__((ext_vector_type(16))) _Float16 v16h;
typedef __attribute__((ext_vector_type(8)))  _Float16 v8h;
typedef __attribute__((ext_vector_type(2)))  _Float16 v2h;
typedef __attribute__((ext_vector_type(8)))  float    v8f;

// ---------------- fragment load (16-bit A/B layout, 16x16x32) ----------------
// Per CDNA5 ISA 7.12.2: lane L holds row M = L&15; kbase = (L>=16) ? 8 : 0.
// v16h elems 0..7  = src[row][k0+kbase+0 .. +7]
// v16h elems 8..15 = src[row][k0+kbase+16 .. +23]
__device__ __forceinline__ v16h load_frag16(const _Float16* p) {
  v8h lo = *(const v8h*)(p);
  v8h hi = *(const v8h*)(p + 16);
  return __builtin_shufflevector(lo, hi, 0,1,2,3,4,5,6,7,8,9,10,11,12,13,14,15);
}

// 32x32 output tile per wave: C[m0..m0+31][n0..n0+31] += A[M,K] * Bw[N,K]^T
__device__ __forceinline__ void gemm_tile_32x32(
    const _Float16* __restrict__ A, const _Float16* __restrict__ Bw,
    int K, int m0, int n0, int lane, v8f c[4]) {
  const int kb = (lane >> 4) << 3;      // 0 or 8
  const _Float16* pa0 = A  + (size_t)(m0 + (lane & 15)) * K + kb;
  const _Float16* pa1 = pa0 + (size_t)16 * K;
  const _Float16* pb0 = Bw + (size_t)(n0 + (lane & 15)) * K + kb;
  const _Float16* pb1 = pb0 + (size_t)16 * K;
  for (int k = 0; k < K; k += 32) {
    __builtin_prefetch(pa0 + k + 64, 0, 1);
    __builtin_prefetch(pb0 + k + 64, 0, 1);
    v16h a0 = load_frag16(pa0 + k);
    v16h a1 = load_frag16(pa1 + k);
    v16h b0 = load_frag16(pb0 + k);
    v16h b1 = load_frag16(pb1 + k);
    c[0] = __builtin_amdgcn_wmma_f32_16x16x32_f16(false, a0, false, b0, (short)0, c[0], false, false);
    c[1] = __builtin_amdgcn_wmma_f32_16x16x32_f16(false, a0, false, b1, (short)0, c[1], false, false);
    c[2] = __builtin_amdgcn_wmma_f32_16x16x32_f16(false, a1, false, b0, (short)0, c[2], false, false);
    c[3] = __builtin_amdgcn_wmma_f32_16x16x32_f16(false, a1, false, b1, (short)0, c[3], false, false);
  }
}

// C/D layout: lane 0-15 -> N=lane, M=r; lane 16-31 -> N=lane-16, M=r+8
// Variadic so commas inside the epilogue body don't split macro arguments.
#define EPILOGUE_LOOP(...)                                    \
  const int rbase = (lane >> 4) << 3;                         \
  const int cl = lane & 15;                                   \
  for (int t = 0; t < 4; ++t) {                               \
    const int mi = (t >> 1) * 16, ni = (t & 1) * 16;          \
    for (int r = 0; r < 8; ++r) {                             \
      const int row = m0 + mi + rbase + r;                    \
      const int col = n0 + ni + cl;                           \
      const float val = c[t][r];                              \
      __VA_ARGS__                                             \
    }                                                         \
  }

// ---------------- GEMM kernels ----------------
// value projection -> permuted f16 layout [B][H][Nv][Dh]
__global__ void __launch_bounds__(256) gemm_v_kernel(
    const _Float16* __restrict__ A, const _Float16* __restrict__ Bw,
    const float* __restrict__ bias, _Float16* __restrict__ vout,
    int M, int N, int K) {
  const int lane = threadIdx.x & 31, wv = threadIdx.x >> 5;
  const int m0 = blockIdx.x * 32, n0 = blockIdx.y * 256 + wv * 32;
  if (n0 >= N) return;
  v8f c[4] = {};
  gemm_tile_32x32(A, Bw, K, m0, n0, lane, c);
  EPILOGUE_LOOP(
    const int b = row / NVTOT;
    const int nv = row - b * NVTOT;
    const int h = col >> 6;
    const int dh = col & 63;
    vout[(((size_t)(b * HEADS + h)) * NVTOT + nv) * DH + dh] = (_Float16)(val + bias[col]);
  )
}

// plain f32 output (offsets + attention logits)
__global__ void __launch_bounds__(256) gemm_plain_kernel(
    const _Float16* __restrict__ A, const _Float16* __restrict__ Bw,
    const float* __restrict__ bias, float* __restrict__ out,
    int M, int N, int K) {
  const int lane = threadIdx.x & 31, wv = threadIdx.x >> 5;
  const int m0 = blockIdx.x * 32, n0 = blockIdx.y * 256 + wv * 32;
  if (n0 >= N) return;
  v8f c[4] = {};
  gemm_tile_32x32(A, Bw, K, m0, n0, lane, c);
  EPILOGUE_LOOP(
    out[(size_t)row * N + col] = val + bias[col];
  )
}

// residual add epilogue (out-proj and FFN-2)
__global__ void __launch_bounds__(256) gemm_resid_kernel(
    const _Float16* __restrict__ A, const _Float16* __restrict__ Bw,
    const float* __restrict__ bias, const float* __restrict__ resid,
    float* __restrict__ out, int M, int N, int K) {
  const int lane = threadIdx.x & 31, wv = threadIdx.x >> 5;
  const int m0 = blockIdx.x * 32, n0 = blockIdx.y * 256 + wv * 32;
  if (n0 >= N) return;
  v8f c[4] = {};
  gemm_tile_32x32(A, Bw, K, m0, n0, lane, c);
  EPILOGUE_LOOP(
    const size_t idx = (size_t)row * N + col;
    out[idx] = resid[idx] + val + bias[col];
  )
}

// GELU (tanh approx, jax.nn.gelu default) -> f16 activations
__global__ void __launch_bounds__(256) gemm_gelu_kernel(
    const _Float16* __restrict__ A, const _Float16* __restrict__ Bw,
    const float* __restrict__ bias, _Float16* __restrict__ out,
    int M, int N, int K) {
  const int lane = threadIdx.x & 31, wv = threadIdx.x >> 5;
  const int m0 = blockIdx.x * 32, n0 = blockIdx.y * 256 + wv * 32;
  if (n0 >= N) return;
  v8f c[4] = {};
  gemm_tile_32x32(A, Bw, K, m0, n0, lane, c);
  EPILOGUE_LOOP(
    const float x = val + bias[col];
    const float g = 0.5f * x * (1.0f + tanhf(0.7978845608028654f * (x + 0.044715f * x * x * x)));
    out[(size_t)row * N + col] = (_Float16)g;
  )
}

// ---------------- LayerNorm (wave per 384-elem row) ----------------
__global__ void __launch_bounds__(256) ln_kernel(
    const float* __restrict__ src, const float* __restrict__ g,
    const float* __restrict__ bt, float* __restrict__ dst32,
    _Float16* __restrict__ dst16, int M) {
  const int w = (blockIdx.x * blockDim.x + threadIdx.x) >> 5;
  if (w >= M) return;
  const int lane = threadIdx.x & 31;
  const float* row = src + (size_t)w * DIMC;
  float x[12], s = 0.f, s2 = 0.f;
  for (int i = 0; i < 12; ++i) { float v = row[lane + 32 * i]; x[i] = v; s += v; s2 += v * v; }
  for (int o = 16; o; o >>= 1) { s += __shfl_xor(s, o, 32); s2 += __shfl_xor(s2, o, 32); }
  const float mean = s * (1.f / DIMC);
  const float rinv = rsqrtf(s2 * (1.f / DIMC) - mean * mean + 1e-6f);
  for (int i = 0; i < 12; ++i) {
    const int col = lane + 32 * i;
    const float y = (x[i] - mean) * rinv * g[col] + bt[col];
    if (dst32) dst32[(size_t)w * DIMC + col] = y;
    if (dst16) dst16[(size_t)w * DIMC + col] = (_Float16)y;
  }
}

// final: out = query + gamma * LN_n2(y)
__global__ void __launch_bounds__(256) final_kernel(
    const float* __restrict__ src, const float* __restrict__ g,
    const float* __restrict__ bt, const float* __restrict__ gamma,
    const float* __restrict__ query, float* __restrict__ out, int M) {
  const int w = (blockIdx.x * blockDim.x + threadIdx.x) >> 5;
  if (w >= M) return;
  const int lane = threadIdx.x & 31;
  const float* row = src + (size_t)w * DIMC;
  float x[12], s = 0.f, s2 = 0.f;
  for (int i = 0; i < 12; ++i) { float v = row[lane + 32 * i]; x[i] = v; s += v; s2 += v * v; }
  for (int o = 16; o; o >>= 1) { s += __shfl_xor(s, o, 32); s2 += __shfl_xor(s2, o, 32); }
  const float mean = s * (1.f / DIMC);
  const float rinv = rsqrtf(s2 * (1.f / DIMC) - mean * mean + 1e-6f);
  for (int i = 0; i < 12; ++i) {
    const int col = lane + 32 * i;
    const float y = (x[i] - mean) * rinv * g[col] + bt[col];
    out[(size_t)w * DIMC + col] = query[(size_t)w * DIMC + col] + gamma[col] * y;
  }
}

// ---------------- weight prep: f32 -> f16 (+ fused/padded off|aw matrix) ----------------
__global__ void __launch_bounds__(256) prep_weights_kernel(
    const float* __restrict__ Wv, const float* __restrict__ Wo,
    const float* __restrict__ W1, const float* __restrict__ W2,
    const float* __restrict__ Woff, const float* __restrict__ boff,
    const float* __restrict__ Waw, const float* __restrict__ baw,
    _Float16* wv, _Float16* wo, _Float16* w1, _Float16* w2,
    _Float16* woa, float* boa) {
  const int t = blockIdx.x * blockDim.x + threadIdx.x;
  if (t < DIMC * DIMC) { wv[t] = (_Float16)Wv[t]; wo[t] = (_Float16)Wo[t]; }
  if (t < FFN * DIMC)  { w1[t] = (_Float16)W1[t]; w2[t] = (_Float16)W2[t]; }
  if (t < NOA * DIMC) {
    const int r = t / DIMC, k = t - r * DIMC;
    float v = 0.f;
    if (r < 144) v = Woff[r * DIMC + k];
    else if (r < 216) v = Waw[(r - 144) * DIMC + k];
    woa[t] = (_Float16)v;
  }
  if (t < NOA) {
    float v = 0.f;
    if (t < 144) v = boff[t];
    else if (t < 216) v = baw[t - 144];
    boa[t] = v;
  }
}

// ---------------- MSDA sampling: wave per (b,q,head), lane owns 2 channels ----------------
__global__ void __launch_bounds__(256) msda_sample_kernel(
    const float* __restrict__ offaw, const float* __restrict__ refp,
    const _Float16* __restrict__ vbuf, _Float16* __restrict__ sampled) {
  const int gw = (blockIdx.x * blockDim.x + threadIdx.x) >> 5;
  if (gw >= BC * NQC * HEADS) return;
  const int lane = threadIdx.x & 31;
  const int h = gw % HEADS;
  const int q = (gw / HEADS) % NQC;
  const int b = gw / (HEADS * NQC);
  const size_t row = (size_t)b * NQC + q;
  const float* oa = offaw + row * NOA;

  // softmax over the 12 (L*P) logits of this head
  float aw[12], mx = -1e30f;
  #pragma unroll
  for (int j = 0; j < 12; ++j) { aw[j] = oa[144 + h * 12 + j]; mx = fmaxf(mx, aw[j]); }
  float sum = 0.f;
  #pragma unroll
  for (int j = 0; j < 12; ++j) { aw[j] = expf(aw[j] - mx); sum += aw[j]; }
  const float inv = 1.f / sum;

  const int dh0 = lane * 2;
  float acc0 = 0.f, acc1 = 0.f;
  const _Float16* vh = vbuf + (size_t)(b * HEADS + h) * NVTOT * DH;

  const int LW_[3] = {128, 64, 32};
  const int LH_[3] = {128, 64, 32};
  const int LS_[3] = {0, 16384, 20480};
  #pragma unroll
  for (int l = 0; l < LEVELS; ++l) {
    const int Wl = LW_[l], Hl = LH_[l], St = LS_[l];
    const float rx = refp[(row * LEVELS + l) * 2 + 0];
    const float ry = refp[(row * LEVELS + l) * 2 + 1];
    #pragma unroll
    for (int p = 0; p < POINTS; ++p) {
      const float ox = oa[((h * LEVELS + l) * POINTS + p) * 2 + 0];
      const float oy = oa[((h * LEVELS + l) * POINTS + p) * 2 + 1];
      const float x = (rx + ox / (float)Wl) * (float)Wl - 0.5f;   // grid_sample coord
      const float y = (ry + oy / (float)Hl) * (float)Hl - 0.5f;
      const float x0 = floorf(x), y0 = floorf(y);
      const float fx = x - x0, fy = y - y0;
      const int ix = (int)x0, iy = (int)y0;
      const float s = aw[l * POINTS + p] * inv;
      const float w00 = (1.f - fx) * (1.f - fy), w10 = fx * (1.f - fy);
      const float w01 = (1.f - fx) * fy,          w11 = fx * fy;
      #define SAMP(XI, YI, WW)                                                   \
        if ((XI) >= 0 && (XI) < Wl && (YI) >= 0 && (YI) < Hl) {                  \
          const _Float16* vp = vh + ((size_t)(St + (YI) * Wl + (XI))) * DH + dh0;\
          v2h vv = *(const v2h*)vp;                                              \
          acc0 += (WW) * s * (float)vv[0];                                       \
          acc1 += (WW) * s * (float)vv[1];                                       \
        }
      SAMP(ix, iy, w00) SAMP(ix + 1, iy, w10) SAMP(ix, iy + 1, w01) SAMP(ix + 1, iy + 1, w11)
      #undef SAMP
    }
  }
  _Float16* sp = sampled + row * DIMC + h * DH + dh0;
  sp[0] = (_Float16)acc0;
  sp[1] = (_Float16)acc1;
}

// ---------------- host side ----------------
extern "C" void kernel_launch(void* const* d_in, const int* in_sizes, int n_in,
                              void* d_out, int out_size, void* d_ws, size_t ws_size,
                              hipStream_t stream) {
  (void)in_sizes; (void)n_in; (void)out_size; (void)ws_size;
  const float* query = (const float*)d_in[0];
  const float* refp  = (const float*)d_in[1];
  const float* feat  = (const float*)d_in[2];
  // d_in[3]/d_in[4]: spatial shapes / level starts are compile-time constants here
  const float* qn_g = (const float*)d_in[5];
  const float* qn_b = (const float*)d_in[6];
  const float* fn_g = (const float*)d_in[7];
  const float* fn_b = (const float*)d_in[8];
  const float* W_off = (const float*)d_in[9];
  const float* b_off = (const float*)d_in[10];
  const float* W_aw  = (const float*)d_in[11];
  const float* b_aw  = (const float*)d_in[12];
  const float* W_v   = (const float*)d_in[13];
  const float* b_v   = (const float*)d_in[14];
  const float* W_o   = (const float*)d_in[15];
  const float* b_o   = (const float*)d_in[16];
  const float* n1_g  = (const float*)d_in[17];
  const float* n1_b  = (const float*)d_in[18];
  const float* W1    = (const float*)d_in[19];
  const float* b1    = (const float*)d_in[20];
  const float* W2    = (const float*)d_in[21];
  const float* b2    = (const float*)d_in[22];
  const float* n2_g  = (const float*)d_in[23];
  const float* n2_b  = (const float*)d_in[24];
  const float* gamma = (const float*)d_in[25];
  float* out = (float*)d_out;

  // ---- workspace bump allocator (regions reused across pipeline phases) ----
  char* ws = (char*)d_ws;
  size_t off = 0;
  auto take = [&](size_t bytes) -> char* {
    char* p = ws + off;
    off = (off + bytes + 255) & ~(size_t)255;
    return p;
  };
  char* R1 = take((size_t)MV * DIMC * 2);           // 66.0 MB: fn16 -> xres / hbuf
  char* R2 = take((size_t)MV * DIMC * 2);           // 66.0 MB: vbuf -> ybuf
  char* QA = take((size_t)MQ * DIMC * 4);           // 25.2 MB: qn32 -> x1_32
  char* QB = take((size_t)MQ * DIMC * 2);           // 12.6 MB: qn16 -> x1_16
  float*    offaw   = (float*)take((size_t)MQ * NOA * 4);
  _Float16* sampled = (_Float16*)take((size_t)MQ * DIMC * 2);
  _Float16* wv16  = (_Float16*)take((size_t)DIMC * DIMC * 2);
  _Float16* wo16  = (_Float16*)take((size_t)DIMC * DIMC * 2);
  _Float16* w116  = (_Float16*)take((size_t)FFN * DIMC * 2);
  _Float16* w216  = (_Float16*)take((size_t)DIMC * FFN * 2);
  _Float16* woa16 = (_Float16*)take((size_t)NOA * DIMC * 2);
  float*    boa   = (float*)take((size_t)NOA * 4);

  _Float16* fn16 = (_Float16*)R1;
  float*    xres = (float*)R1;
  _Float16* hbuf = (_Float16*)(R1 + (size_t)MQ * DIMC * 4);
  _Float16* vbuf = (_Float16*)R2;
  float*    ybuf = (float*)R2;
  float*    qn32 = (float*)QA;  float*    x132 = (float*)QA;
  _Float16* qn16 = (_Float16*)QB; _Float16* x116 = (_Float16*)QB;

  // 1) weights -> f16 (+ fused padded off|aw matrix)
  prep_weights_kernel<<<(FFN * DIMC + 255) / 256, 256, 0, stream>>>(
      W_v, W_o, W1, W2, W_off, b_off, W_aw, b_aw, wv16, wo16, w116, w216, woa16, boa);
  // 2) LN(feat) -> fn16 ; LN(query) -> qn32 + qn16
  ln_kernel<<<(MV * 32) / 256, 256, 0, stream>>>(feat, fn_g, fn_b, nullptr, fn16, MV);
  ln_kernel<<<(MQ * 32) / 256, 256, 0, stream>>>(query, qn_g, qn_b, qn32, qn16, MQ);
  // 3) value projection (WMMA) -> head-major f16 value tensor
  gemm_v_kernel<<<dim3(MV / 32, (DIMC + 255) / 256), 256, 0, stream>>>(
      fn16, wv16, b_v, vbuf, MV, DIMC, DIMC);
  // 4) offsets + attention logits (WMMA, fused 224-col GEMM)
  gemm_plain_kernel<<<dim3(MQ / 32, 1), 256, 0, stream>>>(
      qn16, woa16, boa, offaw, MQ, NOA, DIMC);
  // 5) softmax + bilinear sampling + attention-weighted accumulation
  msda_sample_kernel<<<(BC * NQC * HEADS * 32) / 256, 256, 0, stream>>>(
      offaw, refp, vbuf, sampled);
  // 6) output projection + identity(qn) residual
  gemm_resid_kernel<<<dim3(MQ / 32, (DIMC + 255) / 256), 256, 0, stream>>>(
      sampled, wo16, b_o, qn32, xres, MQ, DIMC, DIMC);
  // 7) LN n1 -> x1 (f32 + f16)
  ln_kernel<<<(MQ * 32) / 256, 256, 0, stream>>>(xres, n1_g, n1_b, x132, x116, MQ);
  // 8) FFN up + GELU (WMMA)
  gemm_gelu_kernel<<<dim3(MQ / 32, FFN / 256), 256, 0, stream>>>(
      x116, w116, b1, hbuf, MQ, FFN, DIMC);
  // 9) FFN down + residual (WMMA)
  gemm_resid_kernel<<<dim3(MQ / 32, (DIMC + 255) / 256), 256, 0, stream>>>(
      hbuf, w216, b2, x132, ybuf, MQ, DIMC, FFN);
  // 10) LN n2, gamma scale, query residual -> out
  final_kernel<<<(MQ * 32) / 256, 256, 0, stream>>>(
      ybuf, n2_g, n2_b, gamma, query, out, MQ);
}
